// AttenCross_47974784696908
// MI455X (gfx1250) — matrix operand order repaired and checked by reading, not compiled, not run on hardware
//
#include <hip/hip_runtime.h>
#include <hip/hip_bf16.h>

// MI455X (gfx1250, wave32) fused masked-softmax cross-attention score:
//   out[b] = sum_q ( sum_k softmax_k(Q·D^T/sqrt(128), mask) * sim[b,q,k] )
// B=16, QL=1024, DL=4096, H=128.
// One wave32 owns a 16-query tile; GEMM via v_wmma_f32_16x16x32_bf16,
// softmax+sim-weighted reduction fused in registers (scores never hit HBM).

#define B_  16
#define QL_ 1024
#define DL_ 4096
#define H_  128

typedef __bf16 bf16;
typedef __attribute__((ext_vector_type(16))) __bf16 v16bf;
typedef __attribute__((ext_vector_type(8)))  float  v8f;

// Convert 8 consecutive fp32 (16B-aligned) into bf16 elements [E0..E0+7] of a v16bf.
template <int E0>
__device__ __forceinline__ void cvt8(v16bf& v, const float* __restrict__ p) {
    float4 x0 = ((const float4*)p)[0];
    float4 x1 = ((const float4*)p)[1];
    v[E0 + 0] = (bf16)x0.x; v[E0 + 1] = (bf16)x0.y;
    v[E0 + 2] = (bf16)x0.z; v[E0 + 3] = (bf16)x0.w;
    v[E0 + 4] = (bf16)x1.x; v[E0 + 5] = (bf16)x1.y;
    v[E0 + 6] = (bf16)x1.z; v[E0 + 7] = (bf16)x1.w;
}

__global__ __launch_bounds__(256) void atten_fused_kernel(
    const float* __restrict__ qin,   // (B, QL, H)
    const float* __restrict__ din,   // (B, DL, H)
    const int*   __restrict__ dmask, // (B, DL)
    const float* __restrict__ sim,   // (B, QL, DL)
    float*       __restrict__ qval)  // (B, QL) per-query results
{
    const float scale = 0.08838834764831845f; // 1/sqrt(128)

    const int gwave = (blockIdx.x * blockDim.x + threadIdx.x) >> 5; // tile id, 0..1023
    const int lane  = threadIdx.x & 31;
    const int b     = gwave >> 6;          // 64 q-tiles per batch
    const int q0    = (gwave & 63) << 4;   // first query row of this tile

    const int mn   = lane & 15;            // row (A) / column (B,C) index
    const int half = lane >> 4;            // K-half selector per WMMA layouts

    // ---- A fragments: 16x128 query tile as 4x v16bf (bf16), loaded once ----
    // A 16x32 bf16 layout: lane m=mn, element e: K = 16*half + e (contiguous 16).
    const float* qrow  = qin + (size_t)(b * QL_ + q0 + mn) * H_;
    const int    abase = half * 16;
    v16bf a[4];
#pragma unroll
    for (int j = 0; j < 4; ++j) {
        cvt8<0>(a[j], qrow + 32 * j + abase);
        cvt8<8>(a[j], qrow + 32 * j + 8 + abase);
    }

    // Per-lane running sums: num[r] = sum_k e*sim, den[r] = sum_k e, row = r + 8*half.
    float num[8], den[8];
#pragma unroll
    for (int r = 0; r < 8; ++r) { num[r] = 0.0f; den[r] = 0.0f; }

    // B 32x16 bf16 layout: lane col n=mn, element e: K = 16*half + e (contiguous 16).
    const float* drow0  = din + (size_t)(b * DL_ + mn) * H_ + half * 16;
    const int*   mrow   = dmask + b * DL_ + mn;
    const float* simrow = sim + (size_t)(b * QL_ + q0 + 8 * half) * DL_ + mn;

    for (int n0 = 0; n0 < DL_; n0 += 16) {
        const float* drow = drow0 + (size_t)n0 * H_;

        // 16x16 score tile: 4 chained WMMAs over the 128-deep contraction.
        v8f c = {};
#pragma unroll
        for (int j = 0; j < 4; ++j) {
            v16bf bb;
            cvt8<0>(bb, drow + 32 * j);
            cvt8<8>(bb, drow + 32 * j + 8);
            c = __builtin_amdgcn_wmma_f32_16x16x32_bf16(
                    false, a[j], false, bb, (short)0, c, false, false);
        }

        // Masked exp + sim-weighted accumulation (scores ~N(0,1): no max-sub needed).
        const float mk = (mrow[n0] != 0) ? 1.0f : 0.0f;
#pragma unroll
        for (int r = 0; r < 8; ++r) {
            const float e  = mk * __expf(c[r] * scale);
            const float sv = simrow[(size_t)r * DL_ + n0];
            num[r] = fmaf(e, sv, num[r]);
            den[r] += e;
        }
    }

    // Reduce across the 16 lanes of each half (rows r+8*half), write per-query value.
#pragma unroll
    for (int r = 0; r < 8; ++r) {
        float nu = num[r], de = den[r];
#pragma unroll
        for (int off = 1; off < 16; off <<= 1) {
            nu += __shfl_xor(nu, off, 32);
            de += __shfl_xor(de, off, 32);
        }
        if (mn == 0) qval[b * QL_ + q0 + 8 * half + r] = nu / de;
    }
}

// Deterministic per-batch tree reduction of 1024 per-query values -> (B,1).
__global__ __launch_bounds__(256) void reduce_q_kernel(
    const float* __restrict__ qval, float* __restrict__ out)
{
    __shared__ float sm[256];
    const int b = blockIdx.x, t = threadIdx.x;
    float s = 0.0f;
    for (int i = t; i < QL_; i += 256) s += qval[b * QL_ + i];
    sm[t] = s;
    __syncthreads();
    for (int stride = 128; stride > 0; stride >>= 1) {
        if (t < stride) sm[t] += sm[t + stride];
        __syncthreads();
    }
    if (t == 0) out[b] = sm[0];
}

extern "C" void kernel_launch(void* const* d_in, const int* in_sizes, int n_in,
                              void* d_out, int out_size, void* d_ws, size_t ws_size,
                              hipStream_t stream) {
    const float* qin   = (const float*)d_in[0]; // query_input (B,QL,H)
    // d_in[1] = query_mask: unused by the reference computation
    const float* din   = (const float*)d_in[2]; // doc_input (B,DL,H)
    const int*   dmask = (const int*)d_in[3];   // doc_mask (B,DL)
    const float* sim   = (const float*)d_in[4]; // sim_matrix (B,QL,DL)

    float* qval = (float*)d_ws;   // (B,QL) = 64 KB scratch
    float* out  = (float*)d_out;  // (B,1)

    // 1024 q-tiles -> 1024 waves -> 128 blocks of 256 threads (8 waves/WGP pass).
    atten_fused_kernel<<<128, 256, 0, stream>>>(qin, din, dmask, sim, qval);
    reduce_q_kernel<<<B_, 256, 0, stream>>>(qval, out);
}
